// MSACrossAttention_89661737271428
// MI455X (gfx1250) — matrix operand order repaired and checked by reading, not compiled, not run on hardware
//
#include <hip/hip_runtime.h>

#define R_DIM 64
#define L_DIM 512
#define B_DIM 4
#define E_DIM 768
#define H_DIM 12
#define D_DIM 64
#define QD_DIM 384

typedef __attribute__((ext_vector_type(16))) __bf16 v16bf;
typedef __attribute__((ext_vector_type(8)))  __bf16 v8bf;
typedef __attribute__((ext_vector_type(8)))  float  v8f;
typedef unsigned int u32x4 __attribute__((ext_vector_type(4)));
typedef int          i32x4 __attribute__((ext_vector_type(4)));
typedef int          i32x8 __attribute__((ext_vector_type(8)));

union BF16Frag { v16bf v; v8bf h[2]; };

// ---------------- xsum over R: xsum[i*B+n][e] = sum_r x[r,i,n,e] ----------------
__global__ void xsum_kernel(const float* __restrict__ x, float* __restrict__ xsum) {
    const size_t idx = (size_t)blockIdx.x * 256 + threadIdx.x;   // < L*B*E
    const size_t stride = (size_t)L_DIM * B_DIM * E_DIM;
    float s = 0.f;
    #pragma unroll 8
    for (int r = 0; r < R_DIM; ++r) s += x[(size_t)r * stride + idx];
    xsum[idx] = s;
}

// ---------------- fp32 -> bf16 conversion (weights, done once) ------------------
__global__ void f32_to_bf16_kernel(const float* __restrict__ src, __bf16* __restrict__ dst) {
    const size_t i = (size_t)blockIdx.x * 256 + threadIdx.x;
    dst[i] = (__bf16)src[i];
}

// ---------------- softmax over last dim (512), in place, one block per row ------
__global__ void softmax512_kernel(float* __restrict__ p) {
    float* row = p + (size_t)blockIdx.x * 512;
    const int t = threadIdx.x;             // 256 threads, 2 elems each
    float a = row[t], b = row[t + 256];
    __shared__ float red[256];
    red[t] = fmaxf(a, b); __syncthreads();
    for (int s = 128; s > 0; s >>= 1) { if (t < s) red[t] = fmaxf(red[t], red[t + s]); __syncthreads(); }
    const float rmax = red[0]; __syncthreads();
    const float ea = __expf(a - rmax), eb = __expf(b - rmax);
    red[t] = ea + eb; __syncthreads();
    for (int s = 128; s > 0; s >>= 1) { if (t < s) red[t] += red[t + s]; __syncthreads(); }
    const float inv = 1.0f / red[0];
    row[t] = ea * inv; row[t + 256] = eb * inv;
}

// ---------------- TDM: DMA one [tile_rows x 32] bf16 tile into padded LDS -------
// D# per ISA 8.3-8.5: count=1, type=2(image), data_size=1(2B), pad 4 DWORDs after
// every 16 DWORDs (row of 32 bf16 -> LDS row stride 40 bf16). Groups 2/3 zero (2D).
__device__ __forceinline__ void tdm_load_tile_bf16(unsigned lds_byte_addr,
                                                   const __bf16* gptr,
                                                   unsigned stride_elems,
                                                   unsigned tile_rows)
{
    const unsigned long long ga = (unsigned long long)(size_t)gptr;
    u32x4 g0;
    g0[0] = 1u;                                      // count=1 descriptor, user mode
    g0[1] = lds_byte_addr;                           // lds_addr [63:32]
    g0[2] = (unsigned)ga;                            // global_addr [95:64]
    g0[3] = (unsigned)(ga >> 32) | (2u << 30);       // global_addr [120:96] | type=2
    i32x8 g1;
    g1[0] = (1 << 16) | (1 << 20) | (3 << 22) | (3 << 25); // data_size=2B, pad_en, int=16dw, amt=4dw
    g1[1] = 0;                                       // atomic_barrier=0, tensor_dim0[15:0]=0
    g1[2] = 0x4000;                                  // tensor_dim0=0x40000000 (no clamp)
    g1[3] = 0x4000 | (32 << 16);                     // tensor_dim1=0x40000000, tile_dim0=32
    g1[4] = (int)tile_rows;                          // tile_dim1, tile_dim2=0
    g1[5] = (int)stride_elems;                       // tensor_dim0_stride[31:0]
    g1[6] = 0;                                       // stride hi, dim1_stride lo
    g1[7] = 0;
    i32x4 gz = {0, 0, 0, 0};
#if defined(__clang_major__) && __clang_major__ >= 23
    i32x8 gz8 = {0, 0, 0, 0, 0, 0, 0, 0};
    __builtin_amdgcn_tensor_load_to_lds(g0, g1, gz, gz, gz8, 0);
#else
    __builtin_amdgcn_tensor_load_to_lds(g0, g1, gz, gz, 0);
#endif
}

// ---------------- generic batched WMMA GEMM -------------------------------------
// AMODE:  0 = A is fp32 (per-thread stage + cvt), 1 = A is bf16 (TDM DMA stage)
// WMODE:  1 = W is bf16 (TDM stage when TRANSB==0; per-thread transpose when 1)
// TRANSB: 0: C[m,n] = alpha*sum_k A[m,k]*W[n,k] + beta*bias[n]   (W [N,K])
//         1: C[m,n] = alpha*sum_k A[m,k]*W[k,n]                  (W [K,N])
// OUTBF16: C element type (0 fp32, 1 bf16). Offsets are in elements.
// Batch decode: z -> r=z/(zH*zB), h=(z/zB)%zH, n=z%zB.
// Block: 256 threads (8 wave32). Tile 128(M) x 64(N), K-step 32. bf16 WMMA, f32 acc.
template<int AMODE, int WMODE, int TRANSB, int OUTBF16>
__global__ void __launch_bounds__(256)
gemm_wmma(const void* __restrict__ Av, int lda,
          const void* __restrict__ Wv, int ldw,
          const float* __restrict__ bias,
          void* __restrict__ Cv, int ldc,
          int K, float alpha, float beta,
          int zH, int zB,
          int offAr, int offAh, int offAn,
          int offWr, int offWh, int offWn,
          int offCr, int offCh, int offCn)
{
    const int z   = blockIdx.z;
    const int r   = z / (zH * zB);
    const int rem = z - r * (zH * zB);
    const int h   = rem / zB;
    const int nn  = rem - h * zB;
    const size_t aoff = (size_t)r * offAr + (size_t)h * offAh + (size_t)nn * offAn;
    const size_t woff = (size_t)r * offWr + (size_t)h * offWh + (size_t)nn * offWn;
    const size_t coff = (size_t)r * offCr + (size_t)h * offCh + (size_t)nn * offCn;

    const float*  Af = (const float*)Av + aoff;
    const __bf16* Ab = (const __bf16*)Av + aoff;
    const __bf16* Wb = (const __bf16*)Wv + woff;

    const int m0 = blockIdx.y * 128;
    const int n0 = blockIdx.x * 64;

    // padded stride 40 (80B): 16B-aligned b128 LDS reads; matches TDM pad 4dw/16dw
    __shared__ __bf16 lsA[128][40];
    __shared__ __bf16 lsB[64][40];

    const int tid  = threadIdx.x;
    const int wave = tid >> 5;
    const int lane = tid & 31;
    const int lh   = lane & 15;
    const bool hiHalf = lane >= 16;

    const unsigned lsA_addr = (unsigned)(size_t)&lsA[0][0];
    const unsigned lsB_addr = (unsigned)(size_t)&lsB[0][0];

    v8f acc[4];
    #pragma unroll
    for (int t = 0; t < 4; ++t) {
        v8f zv = {0.f, 0.f, 0.f, 0.f, 0.f, 0.f, 0.f, 0.f};
        acc[t] = zv;
    }

    const int arow = tid >> 1;         // 0..127
    const int akb  = (tid & 1) * 16;   // 0,16
    const int vrow = tid >> 3;         // 0..31
    const int vnb  = (tid & 7) * 8;    // 0..56

    const bool useTDM = (AMODE == 1) || (WMODE == 1 && TRANSB == 0);

    for (int k0 = 0; k0 < K; k0 += 32) {
        // ---- TDM DMA staging for bf16 operands (wave 0 issues; EXEC ignored) ----
        if (useTDM && wave == 0) {
            if (AMODE == 1)
                tdm_load_tile_bf16(lsA_addr, Ab + (size_t)m0 * lda + k0, (unsigned)lda, 128);
            if (WMODE == 1 && TRANSB == 0)
                tdm_load_tile_bf16(lsB_addr, Wb + (size_t)n0 * ldw + k0, (unsigned)ldw, 64);
        }
        // ---- per-thread staging: fp32 A tile 128x32 -> bf16 LDS ----
        if (AMODE == 0) {
            const float* ap = Af + (size_t)(m0 + arow) * lda + k0 + akb;
            const float4* ap4 = (const float4*)ap;
            #pragma unroll
            for (int j = 0; j < 4; ++j) {
                float4 f = ap4[j];
                lsA[arow][akb + 4 * j + 0] = (__bf16)f.x;
                lsA[arow][akb + 4 * j + 1] = (__bf16)f.y;
                lsA[arow][akb + 4 * j + 2] = (__bf16)f.z;
                lsA[arow][akb + 4 * j + 3] = (__bf16)f.w;
            }
            if (k0 + 32 < K) __builtin_prefetch(ap + 32, 0, 3);   // global_prefetch_b8
        }
        // ---- per-thread staging: bf16 V [K,N] transposed into lsB[n][k] ----
        if (TRANSB == 1) {
            v8bf raw = *(const v8bf*)(Wb + (size_t)(k0 + vrow) * ldw + n0 + vnb);
            #pragma unroll
            for (int j = 0; j < 8; ++j) lsB[vnb + j][vrow] = raw[j];
        }
        if (useTDM && wave == 0) __builtin_amdgcn_s_wait_tensorcnt(0);
        __syncthreads();

        // ---- fragments per ISA 7.12.2 16-bit layouts, b128 LDS reads ----
        BF16Frag af;
        af.h[0] = *(const v8bf*)&lsA[wave * 16 + lh][hiHalf ? 8 : 0];
        af.h[1] = *(const v8bf*)&lsA[wave * 16 + lh][hiHalf ? 24 : 16];
        const int bk = hiHalf ? 8 : 0;
        #pragma unroll
        for (int t = 0; t < 4; ++t) {
            BF16Frag bf;
            bf.h[0] = *(const v8bf*)&lsB[t * 16 + lh][bk];
            bf.h[1] = *(const v8bf*)&lsB[t * 16 + lh][bk + 16];
            acc[t] = __builtin_amdgcn_wmma_f32_16x16x32_bf16(
                false, af.v, false, bf.v, (short)0, acc[t], false, false);
        }
        __syncthreads();
    }

    // ---- epilogue: D layout lanes<16 -> M=g, lanes>=16 -> M=g+8; N=lane&15 ----
    #pragma unroll
    for (int t = 0; t < 4; ++t) {
        const int nc = n0 + t * 16 + lh;
        const float bterm = bias ? beta * bias[nc] : 0.0f;
        #pragma unroll
        for (int g = 0; g < 8; ++g) {
            const int mc = m0 + wave * 16 + g + (hiHalf ? 8 : 0);
            const float val = alpha * acc[t][g] + bterm;
            if (OUTBF16)
                ((__bf16*)Cv)[coff + (size_t)mc * ldc + nc] = (__bf16)val;
            else
                ((float*)Cv)[coff + (size_t)mc * ldc + nc] = val;
        }
    }
}

template<int AMODE, int WMODE, int TRANSB, int OUTBF16>
static inline void launch_gemm(dim3 grid,
                               const void* A, int lda, const void* W, int ldw,
                               const float* bias, void* C, int ldc,
                               int K, float alpha, float beta,
                               int zH, int zB,
                               int oAr, int oAh, int oAn,
                               int oWr, int oWh, int oWn,
                               int oCr, int oCh, int oCn,
                               hipStream_t s)
{
    gemm_wmma<AMODE, WMODE, TRANSB, OUTBF16><<<grid, dim3(256), 0, s>>>(
        A, lda, W, ldw, bias, C, ldc, K, alpha, beta, zH, zB,
        oAr, oAh, oAn, oWr, oWh, oWn, oCr, oCh, oCn);
}

extern "C" void kernel_launch(void* const* d_in, const int* in_sizes, int n_in,
                              void* d_out, int out_size, void* d_ws, size_t ws_size,
                              hipStream_t stream)
{
    (void)in_sizes; (void)n_in; (void)out_size;
    const float* x     = (const float*)d_in[0];
    const float* query = (const float*)d_in[1];
    const float* Wq = (const float*)d_in[2];
    const float* bq = (const float*)d_in[3];
    const float* Wk = (const float*)d_in[4];
    const float* bk = (const float*)d_in[5];
    const float* Wv = (const float*)d_in[6];
    const float* bv = (const float*)d_in[7];
    const float* Wo = (const float*)d_in[8];
    const float* bo = (const float*)d_in[9];

    float* out  = (float*)d_out;                                  // [R,L,B,E] fp32
    float* attn = out + (size_t)R_DIM * L_DIM * B_DIM * E_DIM;    // [H,B,L,L] fp32

    const size_t slice = (size_t)L_DIM * B_DIM * E_DIM;           // 1,572,864 elems per r
    const size_t wqn = (size_t)E_DIM * QD_DIM;                    // 294,912
    const size_t wen = (size_t)E_DIM * E_DIM;                     // 589,824

    // workspace layout: xsum(f32) | q(bf16) | ks(bf16) | Wq/Wk/Wv/Wo(bf16) | v | ctx
    char* wsb = (char*)d_ws;
    float*  ws_xsum = (float*)wsb;
    __bf16* ws_q    = (__bf16*)(wsb + slice * sizeof(float));
    __bf16* ws_ks   = ws_q + slice;
    __bf16* wWq = ws_ks + slice;
    __bf16* wWk = wWq + wqn;
    __bf16* wWv = wWk + wen;
    __bf16* wWo = wWv + wen;
    __bf16* ws_v = wWo + wen;

    const size_t fixedB = slice * sizeof(float) + 2 * slice * 2 + (wqn + 3 * wen) * 2;
    const size_t chunkB = 2 * slice * 2;                          // v + ctx per r (bf16)
    int CH = 1;
    if (ws_size > fixedB + chunkB) CH = (int)((ws_size - fixedB) / chunkB);
    if (CH > R_DIM) CH = R_DIM;
    if (CH < 1) CH = 1;
    __bf16* ws_ctx = ws_v + (size_t)CH * slice;

    const float scaling = 0.125f / 8.0f;   // D^-0.5 / sqrt(R) = 1/64
    const int BLL = B_DIM * L_DIM * L_DIM, LL = L_DIM * L_DIM;

    // 0) weights -> bf16 (once per call; tiny)
    f32_to_bf16_kernel<<<dim3((unsigned)(wqn / 256)), dim3(256), 0, stream>>>(Wq, wWq);
    f32_to_bf16_kernel<<<dim3((unsigned)(wen / 256)), dim3(256), 0, stream>>>(Wk, wWk);
    f32_to_bf16_kernel<<<dim3((unsigned)(wen / 256)), dim3(256), 0, stream>>>(Wv, wWv);
    f32_to_bf16_kernel<<<dim3((unsigned)(wen / 256)), dim3(256), 0, stream>>>(Wo, wWo);

    // 1) xsum = sum_r x  (exploits linearity of the r-sum in the logits)
    xsum_kernel<<<dim3((unsigned)(slice / 256)), dim3(256), 0, stream>>>(x, ws_xsum);

    // 2) q = query @ Wq^T + bq -> bf16   [2048 x 768], K=384
    launch_gemm<0,1,0,1>(dim3(E_DIM / 64, 2048 / 128, 1),
                query, QD_DIM, wWq, QD_DIM, bq, ws_q, E_DIM,
                QD_DIM, 1.f, 1.f, 1, 1, 0,0,0, 0,0,0, 0,0,0, stream);

    // 3) ks = scaling*(xsum @ Wk^T) + R*scaling*bk -> bf16   [2048 x 768], K=768
    launch_gemm<0,1,0,1>(dim3(E_DIM / 64, 2048 / 128, 1),
                ws_xsum, E_DIM, wWk, E_DIM, bk, ws_ks, E_DIM,
                E_DIM, scaling, (float)R_DIM * scaling, 1, 1, 0,0,0, 0,0,0, 0,0,0, stream);

    // 4) logits[h,n,i,j] = ks_hn @ q_hn^T (K=64), both bf16 via TDM, fp32 out
    launch_gemm<1,1,0,0>(dim3(L_DIM / 64, L_DIM / 128, H_DIM * B_DIM),
                ws_ks, B_DIM * E_DIM, ws_q, B_DIM * E_DIM, nullptr,
                attn, L_DIM, D_DIM, 1.f, 0.f,
                H_DIM, B_DIM,
                0, D_DIM, E_DIM,
                0, D_DIM, E_DIM,
                0, BLL, LL, stream);

    // 5) softmax over j, in place in attn output region (fp32)
    softmax512_kernel<<<dim3(H_DIM * B_DIM * L_DIM), dim3(256), 0, stream>>>(attn);

    // 6) chunked over r: v-proj (bf16 out), context (P @ V, transB), out-proj
    for (int r0 = 0; r0 < R_DIM; r0 += CH) {
        const int ch = (R_DIM - r0 < CH) ? (R_DIM - r0) : CH;
        const float* xr = x + (size_t)r0 * slice;

        // v = x_r @ Wv^T + bv -> bf16   [ch*2048 x 768], K=768
        launch_gemm<0,1,0,1>(dim3(E_DIM / 64, (unsigned)(ch * 2048 / 128), 1),
                    xr, E_DIM, wWv, E_DIM, bv, ws_v, E_DIM,
                    E_DIM, 1.f, 1.f, 1, 1, 0,0,0, 0,0,0, 0,0,0, stream);

        // context[r,i,n,h,d] = sum_j P[h,n,i,j] * v[r,j,n,h,d] -> bf16
        launch_gemm<0,1,1,1>(dim3(1, L_DIM / 128, (unsigned)(ch * H_DIM * B_DIM)),
                    attn, L_DIM, ws_v, B_DIM * E_DIM, nullptr,
                    ws_ctx, B_DIM * E_DIM, L_DIM, 1.f, 0.f,
                    H_DIM, B_DIM,
                    0, BLL, LL,                    // A: (h*B+n)*L*L
                    (int)slice, D_DIM, E_DIM,      // W: r*slice + h*64 + n*768
                    (int)slice, D_DIM, E_DIM,      // C: r*slice + h*64 + n*768
                    stream);

        // out_r = ctx @ Wo^T + bo -> fp32   [ch*2048 x 768], K=768 (A+W via TDM)
        launch_gemm<1,1,0,0>(dim3(E_DIM / 64, (unsigned)(ch * 2048 / 128), 1),
                    ws_ctx, E_DIM, wWo, E_DIM, bo,
                    out + (size_t)r0 * slice, E_DIM,
                    E_DIM, 1.f, 1.f, 1, 1, 0,0,0, 0,0,0, 0,0,0, stream);
    }
}
